// GroupQueryAttention_80410377716194
// MI455X (gfx1250) — matrix operand order repaired and checked by reading, not compiled, not run on hardware
//
#include <hip/hip_runtime.h>
#include <stdint.h>

// ---------------- problem constants ----------------
constexpr int Bc  = 4;
constexpr int Sc  = 2048;
constexpr int Dc  = 2048;
constexpr int NHc = 16;
constexpr int NKVc= 4;
constexpr int HDc = 128;
constexpr int Mrows = Bc * Sc;              // 8192
constexpr float SCALE = 0.08838834764831845f; // 1/sqrt(128)
constexpr int KC = 64;                      // k-chunk per LDS stage in GEMM

// ---------------- vector types ----------------
typedef __attribute__((ext_vector_type(16))) __bf16 v16bf;
typedef __attribute__((ext_vector_type(8)))  __bf16 v8bf;
typedef __attribute__((ext_vector_type(8)))  float  v8f;

__device__ inline uint16_t f2bf(float f) {
  uint32_t u = __float_as_uint(f);
  uint32_t r = u + 0x7FFFu + ((u >> 16) & 1u);      // RNE
  if ((u & 0x7F800000u) == 0x7F800000u) r = u;      // inf / NaN passthrough
  return (uint16_t)(r >> 16);
}

__device__ inline v16bf cat8(v8bf lo, v8bf hi) {
  v16bf r;
#pragma unroll
  for (int i = 0; i < 8; ++i) { r[i] = lo[i]; r[8 + i] = hi[i]; }
  return r;
}

__device__ inline v8f wmma_bf16(v16bf a, v16bf b, v8f c) {
  return __builtin_amdgcn_wmma_f32_16x16x32_bf16(
      false, a, false, b, (short)0, c, false, false);
}

__device__ inline float rmax16(float v) {
#pragma unroll
  for (int o = 8; o > 0; o >>= 1) v = fmaxf(v, __shfl_xor(v, o, 16));
  return v;
}
__device__ inline float rsum16(float v) {
#pragma unroll
  for (int o = 8; o > 0; o >>= 1) v += __shfl_xor(v, o, 16);
  return v;
}

// async 16B copy: global (64-bit VGPR pair) -> LDS (32-bit byte address).
// INST_OFFSET is added to BOTH addresses per the ISA async pseudocode.
__device__ inline void async_g2l_16(uint32_t lds_addr, const void* gaddr) {
  asm volatile("global_load_async_to_lds_b128 %0, %1, off"
               :: "v"(lds_addr), "v"(gaddr) : "memory");
}
__device__ inline void async_g2l_16_off16(uint32_t lds_addr, const void* gaddr) {
  asm volatile("global_load_async_to_lds_b128 %0, %1, off offset:16"
               :: "v"(lds_addr), "v"(gaddr) : "memory");
}
__device__ inline void wait_async0() {
  asm volatile("s_wait_asynccnt 0x0" ::: "memory");
}

// ---------------- conversion kernels ----------------
__global__ void cvt_f32_bf16(const float* __restrict__ src,
                             uint16_t* __restrict__ dst, long n) {
  long i = (long)blockIdx.x * blockDim.x + threadIdx.x;
  if (i < n) dst[i] = f2bf(src[i]);
}

// Wt[n*K + k] = bf16(W[k*N + n])   (weights stored transposed for B-fragments)
__global__ void cvt_transpose_bf16(const float* __restrict__ W,
                                   uint16_t* __restrict__ Wt,
                                   int Kdim, int Ndim) {
  long i = (long)blockIdx.x * blockDim.x + threadIdx.x;
  long total = (long)Kdim * Ndim;
  if (i < total) {
    int nn = (int)(i / Kdim);
    int kk = (int)(i % Kdim);
    Wt[i] = f2bf(W[(long)kk * Ndim + nn]);
  }
}

// ---------------- generic WMMA GEMM ----------------
// C[M,N] = A_bf16[M,K] @ Bt_bf16[N,K]^T + bias
// B tile (shared by all 8 waves of the block) is staged into LDS with
// double-buffered global_load_async_to_lds_b128 (ASYNCcnt); A fragments are
// per-wave unique and stay direct-global. Per stage: all loads are issued
// first (clausable, one wait), then 8 back-to-back WMMAs.
// mode 0: write bf16 to [B,NH,S,HD]     (Q layout)
// mode 1: write bf16 to [B,NKV,S,HD]    (K layout)
// mode 2: write bf16 to [B,NKV,HD,S]    (V^T layout)
// mode 3: write fp32 row-major [M,N]    (final output)
__global__ void __launch_bounds__(256)
gemm_wmma(const uint16_t* __restrict__ A, const uint16_t* __restrict__ Bt,
          const float* __restrict__ bias, void* __restrict__ out,
          int M, int N, int K, int mode) {
  __shared__ __align__(16) uint16_t bstage[2][64 * KC];   // 2 x 8 KB

  const int tid  = threadIdx.x;
  const int lane = tid & 31;
  const int wave = tid >> 5;
  const int m0 = (blockIdx.y * 8 + wave) * 16;
  const int n0 = blockIdx.x * 64;

  const int am   = m0 + (lane & 15);          // A row for this lane
  const int kb8  = (lane & 16) ? 8 : 0;       // unused; A loads use koff
  const int koff = (lane & 16) ? 16 : 0;      // A/B k-offset (contiguous k = 16h + s)

  // staging assignment: thread t copies 32B = elems [cpart*16, cpart*16+16)
  // of B-tile column ccol; 4 consecutive threads cover one 128B line.
  const int ccol  = tid >> 2;
  const int cpart = tid & 3;
  const __bf16* csrc = (const __bf16*)Bt + (long)(n0 + ccol) * K + cpart * 16;
  const uint32_t cdst[2] = {
    (uint32_t)(size_t)&bstage[0][ccol * KC + cpart * 16],
    (uint32_t)(size_t)&bstage[1][ccol * KC + cpart * 16]
  };

  v8f zero = {};
  v8f acc[4] = {zero, zero, zero, zero};
  const __bf16* Arow = (const __bf16*)A + (long)am * K;

  // prologue: stage 0 into buffer 0
  async_g2l_16(cdst[0], csrc);
  async_g2l_16_off16(cdst[0], csrc);

  const int nstage = K / KC;                  // 32 stages of 64 k-values
  for (int st = 0; st < nstage; ++st) {
    wait_async0();                            // my copies for this stage done
    __syncthreads();                          // everyone's copies visible
    if (st + 1 < nstage) {                    // prefetch next stage (other buf)
      const __bf16* s = csrc + (st + 1) * KC;
      uint32_t d = cdst[(st + 1) & 1];
      async_g2l_16(d, s);
      async_g2l_16_off16(d, s);
    }
    const __bf16* buf = (const __bf16*)&bstage[st & 1][0];
    const int k0 = st * KC;
    __builtin_prefetch(Arow + k0 + 2 * KC, 0, 1);

    // issue ALL loads for this stage first, then the WMMA chain
    v16bf af[2];
    v16bf bf[2][4];
#pragma unroll
    for (int j = 0; j < 2; ++j) {             // two 32-k WMMA steps per stage
      v8bf alo = *(const v8bf*)(Arow + k0 + j * 32 + koff);
      v8bf ahi = *(const v8bf*)(Arow + k0 + j * 32 + koff + 8);
      af[j] = cat8(alo, ahi);
    }
#pragma unroll
    for (int j = 0; j < 2; ++j)
#pragma unroll
      for (int t = 0; t < 4; ++t)
        bf[j][t] = *(const v16bf*)(buf + (long)(t * 16 + (lane & 15)) * KC
                                       + j * 32 + koff);
#pragma unroll
    for (int j = 0; j < 2; ++j)
#pragma unroll
      for (int t = 0; t < 4; ++t)
        acc[t] = wmma_bf16(af[j], bf[j][t], acc[t]);
  }

  // epilogue: bias + layout-specific store
  const int hih = (lane & 16) ? 8 : 0;
#pragma unroll
  for (int t = 0; t < 4; ++t) {
    int n = n0 + t * 16 + (lane & 15);
    float bv = bias[n];
#pragma unroll
    for (int r = 0; r < 8; ++r) {
      int m = m0 + r + hih;
      float val = acc[t][r] + bv;
      if (mode == 3) {
        ((float*)out)[(long)m * N + n] = val;
      } else {
        int b_ = m / Sc, s_ = m % Sc;
        int h_ = n / HDc, hd = n % HDc;
        long idx;
        if (mode == 2)
          idx = (((long)b_ * NKVc + h_) * HDc + hd) * Sc + s_;
        else {
          int Hh = (mode == 0) ? NHc : NKVc;
          idx = (((long)b_ * Hh + h_) * Sc + s_) * HDc + hd;
        }
        ((uint16_t*)out)[idx] = f2bf(val);
      }
    }
  }
}

// ---------------- flash attention (bf16 WMMA, fp32 softmax) ----------------
// one wave per (b, h, 16-query tile); 32 keys per inner iteration
__global__ void __launch_bounds__(256)
attn_wmma(const uint16_t* __restrict__ Q, const uint16_t* __restrict__ Kb,
          const uint16_t* __restrict__ VT, uint16_t* __restrict__ AO) {
  __shared__ __align__(16) uint16_t pshare[8 * 16 * 32]; // per-wave 16x32 bf16 P tile

  const int lane = threadIdx.x & 31;
  const int wave = threadIdx.x >> 5;
  const int task = blockIdx.x * 8 + wave;       // 8192 tasks total
  const int qt = task & 127;                    // q tile (S/16 = 128)
  const int h  = (task >> 7) & 15;
  const int b  = task >> 11;
  const int kvh = h >> 2;                       // GQA group = 4

  const __bf16* Qp = (const __bf16*)Q + (((long)(b * NHc + h) * Sc) + qt * 16) * HDc;
  const __bf16* Kp = (const __bf16*)Kb + ((long)(b * NKVc + kvh) * Sc) * HDc;
  const __bf16* Vp = (const __bf16*)VT + ((long)(b * NKVc + kvh) * HDc) * Sc;

  const int mq  = lane & 15;
  const int kb8 = (lane & 16) ? 8 : 0;
  const int koff= (lane & 16) ? 16 : 0;
  const int hih = (lane & 16) ? 8 : 0;
  const int nl  = lane & 15;

  // resident Q fragments: 16 x 128 over 4 K-steps of 32
  v16bf qf[4];
#pragma unroll
  for (int ks = 0; ks < 4; ++ks) {
    v8bf lo = *(const v8bf*)(Qp + (long)mq * HDc + ks * 32 + koff);
    v8bf hi = *(const v8bf*)(Qp + (long)mq * HDc + ks * 32 + koff + 8);
    qf[ks] = cat8(lo, hi);
  }

  v8f zero = {};
  v8f acc[8];
#pragma unroll
  for (int i = 0; i < 8; ++i) acc[i] = zero;
  float mrow[8], lrow[8];
#pragma unroll
  for (int r = 0; r < 8; ++r) { mrow[r] = -3.0e38f; lrow[r] = 0.0f; }

  uint16_t* ps = pshare + wave * 512;
  const int nk = (qt * 16 + 15) / 32 + 1;       // causal: 32-key chunks

  for (int kt = 0; kt < nk; ++kt) {
    const int kb32 = kt * 32;

    // scores: Q(16x128) @ K^T(128x32) -> two 16x16 f32 tiles
    v8f c0 = zero, c1 = zero;
#pragma unroll
    for (int ks = 0; ks < 4; ++ks) {
      v16bf kf0 = *(const v16bf*)(Kp + (long)(kb32 + nl) * HDc + ks * 32 + koff);
      v16bf kf1 = *(const v16bf*)(Kp + (long)(kb32 + 16 + nl) * HDc + ks * 32 + koff);
      c0 = wmma_bf16(qf[ks], kf0, c0);
      c1 = wmma_bf16(qf[ks], kf1, c1);
    }

    // causal mask + online softmax (fp32)
    float p0[8], p1[8], corr[8];
#pragma unroll
    for (int r = 0; r < 8; ++r) {
      int q = qt * 16 + r + hih;
      float s0 = (kb32 + nl      <= q) ? c0[r] * SCALE : -1.0e30f;
      float s1 = (kb32 + 16 + nl <= q) ? c1[r] * SCALE : -1.0e30f;
      float tm = rmax16(fmaxf(s0, s1));
      float nm = fmaxf(mrow[r], tm);
      corr[r] = __expf(mrow[r] - nm);
      p0[r] = __expf(s0 - nm);
      p1[r] = __expf(s1 - nm);
      lrow[r] = lrow[r] * corr[r] + rsum16(p0[r] + p1[r]);
      mrow[r] = nm;
    }
#pragma unroll
    for (int nn = 0; nn < 8; ++nn)
#pragma unroll
      for (int r = 0; r < 8; ++r) acc[nn][r] *= corr[r];

    // P: C-layout -> A-layout through per-wave LDS slice
#pragma unroll
    for (int r = 0; r < 8; ++r) {
      int row = r + hih;
      ps[row * 32 + nl]      = f2bf(p0[r]);
      ps[row * 32 + 16 + nl] = f2bf(p1[r]);
    }
    asm volatile("s_wait_dscnt 0x0" ::: "memory");
    v8bf plo = *(const v8bf*)((const __bf16*)ps + mq * 32 + koff);
    v8bf phi = *(const v8bf*)((const __bf16*)ps + mq * 32 + koff + 8);
    v16bf pf = cat8(plo, phi);

    // O += P(16x32) @ V(32x128): 8 WMMAs over HD n-tiles
#pragma unroll
    for (int nn = 0; nn < 8; ++nn) {
      v16bf vf = *(const v16bf*)(Vp + (long)(nn * 16 + nl) * Sc + kb32 + koff);
      acc[nn] = wmma_bf16(pf, vf, acc[nn]);
    }
  }

  // normalize and write bf16 attn-out as [B*S, NH*HD] row-major (A layout for out-proj)
#pragma unroll
  for (int nn = 0; nn < 8; ++nn) {
#pragma unroll
    for (int r = 0; r < 8; ++r) {
      float val = acc[nn][r] / lrow[r];
      long row = (long)b * Sc + qt * 16 + r + hih;
      int col = h * 128 + nn * 16 + nl;
      AO[row * (long)Dc + col] = f2bf(val);
    }
  }
}

// ---------------- workspace layout (bytes) ----------------
constexpr size_t SZ_XBF = (size_t)Mrows * Dc * 2;          // 32 MB
constexpr size_t SZ_WQT = (size_t)Dc * (NHc * HDc) * 2;    // 8 MB
constexpr size_t SZ_WKT = (size_t)Dc * (NKVc * HDc) * 2;   // 2 MB
constexpr size_t SZ_WVT = SZ_WKT;
constexpr size_t SZ_WOT = (size_t)Dc * Dc * 2;             // 8 MB
constexpr size_t SZ_QBF = (size_t)Bc * NHc * Sc * HDc * 2; // 32 MB
constexpr size_t SZ_KBF = (size_t)Bc * NKVc * Sc * HDc * 2;// 8 MB
constexpr size_t SZ_VTB = SZ_KBF;
constexpr size_t SZ_AOB = (size_t)Mrows * Dc * 2;          // 32 MB

constexpr size_t OFF_XBF = 0;
constexpr size_t OFF_WQT = OFF_XBF + SZ_XBF;
constexpr size_t OFF_WKT = OFF_WQT + SZ_WQT;
constexpr size_t OFF_WVT = OFF_WKT + SZ_WKT;
constexpr size_t OFF_WOT = OFF_WVT + SZ_WVT;
constexpr size_t OFF_QBF = OFF_WOT + SZ_WOT;
constexpr size_t OFF_KBF = OFF_QBF + SZ_QBF;
constexpr size_t OFF_VTB = OFF_KBF + SZ_KBF;
constexpr size_t OFF_AOB = OFF_VTB + SZ_VTB;

extern "C" void kernel_launch(void* const* d_in, const int* in_sizes, int n_in,
                              void* d_out, int out_size, void* d_ws, size_t ws_size,
                              hipStream_t stream) {
  const float* x  = (const float*)d_in[0];
  const float* Wq = (const float*)d_in[1];
  const float* bq = (const float*)d_in[2];
  const float* Wk = (const float*)d_in[3];
  const float* bk = (const float*)d_in[4];
  const float* Wv = (const float*)d_in[5];
  const float* bv = (const float*)d_in[6];
  const float* Wo = (const float*)d_in[7];
  const float* bo = (const float*)d_in[8];

  char* ws = (char*)d_ws;
  uint16_t* xbf = (uint16_t*)(ws + OFF_XBF);
  uint16_t* wqt = (uint16_t*)(ws + OFF_WQT);
  uint16_t* wkt = (uint16_t*)(ws + OFF_WKT);
  uint16_t* wvt = (uint16_t*)(ws + OFF_WVT);
  uint16_t* wot = (uint16_t*)(ws + OFF_WOT);
  uint16_t* qbf = (uint16_t*)(ws + OFF_QBF);
  uint16_t* kbf = (uint16_t*)(ws + OFF_KBF);
  uint16_t* vtb = (uint16_t*)(ws + OFF_VTB);
  uint16_t* aob = (uint16_t*)(ws + OFF_AOB);

  // 1) precision conversion + weight transposes
  {
    long n = (long)Mrows * Dc;
    cvt_f32_bf16<<<(unsigned)((n + 255) / 256), 256, 0, stream>>>(x, xbf, n);
  }
  {
    long n = (long)Dc * (NHc * HDc);
    cvt_transpose_bf16<<<(unsigned)((n + 255) / 256), 256, 0, stream>>>(Wq, wqt, Dc, NHc * HDc);
  }
  {
    long n = (long)Dc * (NKVc * HDc);
    cvt_transpose_bf16<<<(unsigned)((n + 255) / 256), 256, 0, stream>>>(Wk, wkt, Dc, NKVc * HDc);
    cvt_transpose_bf16<<<(unsigned)((n + 255) / 256), 256, 0, stream>>>(Wv, wvt, Dc, NKVc * HDc);
  }
  {
    long n = (long)Dc * Dc;
    cvt_transpose_bf16<<<(unsigned)((n + 255) / 256), 256, 0, stream>>>(Wo, wot, Dc, Dc);
  }

  // 2) QKV projections (WMMA), writing attention-ready bf16 layouts
  {
    dim3 grid(2048 / 64, Mrows / 128);
    gemm_wmma<<<grid, 256, 0, stream>>>(xbf, wqt, bq, qbf, Mrows, 2048, Dc, 0);
  }
  {
    dim3 grid(512 / 64, Mrows / 128);
    gemm_wmma<<<grid, 256, 0, stream>>>(xbf, wkt, bk, kbf, Mrows, 512, Dc, 1);
    gemm_wmma<<<grid, 256, 0, stream>>>(xbf, wvt, bv, vtb, Mrows, 512, Dc, 2);
  }

  // 3) causal GQA flash attention (WMMA)
  attn_wmma<<<Bc * NHc * (Sc / 16) / 8, 256, 0, stream>>>(qbf, kbf, vtb, aob);

  // 4) output projection -> fp32 d_out
  {
    dim3 grid(2048 / 64, Mrows / 128);
    gemm_wmma<<<grid, 256, 0, stream>>>(aob, wot, bo, d_out, Mrows, 2048, Dc, 3);
  }
}
